// Head_81862076662136
// MI455X (gfx1250) — matrix operand order repaired and checked by reading, not compiled, not run on hardware
//
#include <hip/hip_runtime.h>
#include <hip/hip_bf16.h>

typedef __attribute__((ext_vector_type(16))) _Float16 v16h;
typedef __attribute__((ext_vector_type(8)))  float    v8f;

#define BB 16
#define TT 4096
#define DD 64

__device__ __forceinline__ v8f wmma_f16(v16h a, v16h b, v8f c) {
  return __builtin_amdgcn_wmma_f32_16x16x32_f16(false, a, false, b, (short)0, c,
                                                false, false);
}

// ---------------------------------------------------------------------------
// Kernel 1: projections  K = x Wk^T, Q = x Wq^T (row-major f16),
//                        V = x Wv^T stored TRANSPOSED as [D][T] f16.
// One wave computes a 16-row tile of all three projections via WMMA.
// ---------------------------------------------------------------------------
__global__ __launch_bounds__(256) void proj_kernel(
    const float* __restrict__ x,  const float* __restrict__ Wk,
    const float* __restrict__ Wq, const float* __restrict__ Wv,
    _Float16* __restrict__ kp, _Float16* __restrict__ qp,
    _Float16* __restrict__ vt)
{
  const int lane = threadIdx.x & 31;
  const int wid  = threadIdx.x >> 5;
  const int tile = blockIdx.x * 8 + wid;       // 0 .. B*T/16-1
  const int b    = tile >> 8;                  // T/16 == 256 tiles per batch
  const int t0   = (tile & 255) << 4;
  const int lrow = lane & 15;
  const int hi   = lane >> 4;                  // half-wave id
  const int k0   = hi << 3;                    // A-matrix K base (0 or 8)
  const int rb   = hi << 3;                    // C-matrix row base (0 or 8)

  // A = x rows (16 x 64, contraction over channels), fp32 -> fp16 on the fly.
  const float* xr = x + ((size_t)b * TT + (t0 + lrow)) * DD;
  v16h xa0, xa1;
#pragma unroll
  for (int j = 0; j < 8; ++j) {
    xa0[j]     = (_Float16)xr[k0 + j];
    xa0[8 + j] = (_Float16)xr[16 + k0 + j];
    xa1[j]     = (_Float16)xr[32 + k0 + j];
    xa1[8 + j] = (_Float16)xr[48 + k0 + j];
  }

  const float* Ws[3] = {Wk, Wq, Wv};
  for (int p = 0; p < 3; ++p) {
    const float* W = Ws[p];
#pragma unroll
    for (int n0 = 0; n0 < 4; ++n0) {
      // B column h = W row h; lanes<16 hold K=0..7,16..23, lanes>=16 hold K=8..15,24..31.
      const int h = (n0 << 4) + lrow;
      const float* wr = W + (size_t)h * DD + (hi << 3);
      v16h b0, b1;
#pragma unroll
      for (int j = 0; j < 16; ++j) {
        b0[j] = (_Float16)wr[j < 8 ? j : j + 8];
        b1[j] = (_Float16)wr[32 + (j < 8 ? j : j + 8)];
      }
      v8f c = {};
      c = wmma_f16(xa0, b0, c);
      c = wmma_f16(xa1, b1, c);
      // C layout: lane = column h, VGPR v = row rb+v.
      if (p == 2) {  // V transposed: vt[b][d][t]
        _Float16* dst = vt + (size_t)b * DD * TT;
#pragma unroll
        for (int v = 0; v < 8; ++v)
          dst[(size_t)((n0 << 4) + lrow) * TT + (t0 + rb + v)] = (_Float16)c[v];
      } else {       // K / Q row-major: [b][t][h]
        _Float16* dst = (p == 0 ? kp : qp) + (size_t)b * TT * DD;
#pragma unroll
        for (int v = 0; v < 8; ++v)
          dst[(size_t)(t0 + rb + v) * DD + (n0 << 4) + lrow] = (_Float16)c[v];
      }
    }
  }
}

// ---------------------------------------------------------------------------
// Kernel 2: causal flash attention.  scores[t][s] = K[t] . Q[s], softmax over
// s, O = P V.  One wave owns 32 query rows as TWO 16-wide t-groups that share
// every Q-tile and V-tile load (halves memory traffic per flop: 8 loads feed
// 16 WMMAs per 32-key step).  Z is computed transposed (Z^T = Q_s x K_t^T) so
// softmax stats are per-lane + one xor-16 shuffle, and the two 16x16 Z tiles
// of a 32-wide s step ARE the A-matrix layout of P^T (16x32 f16): the PV
// repack needs only f32->f16 converts, zero cross-lane movement.
// ---------------------------------------------------------------------------
__global__ __launch_bounds__(256) void attn_kernel(
    const _Float16* __restrict__ kp, const _Float16* __restrict__ qp,
    const _Float16* __restrict__ vt, float* __restrict__ out)
{
  const int lane = threadIdx.x & 31;
  const int wid  = threadIdx.x >> 5;
  const int tile = blockIdx.x * 8 + wid;       // 0 .. B*T/32-1
  const int b    = tile >> 7;                  // T/32 == 128 tiles per batch
  const int qt0  = (tile & 127) << 5;          // first of 32 query rows
  const int lrow = lane & 15;
  const int hi   = lane >> 4;
  const int k0a  = hi << 3;      // A-layout K base
  const int rb   = hi << 3;      // C-layout row base

  const _Float16* Kb = kp + (size_t)b * TT * DD;
  const _Float16* Qb = qp + (size_t)b * TT * DD;
  const _Float16* Vb = vt + (size_t)b * DD * TT;

  // Score B operands: the wave's 32 K-rows as two 16-column groups (fixed).
  v16h kb[2][2];
#pragma unroll
  for (int g = 0; g < 2; ++g) {
    const _Float16* kr = Kb + (size_t)(qt0 + (g << 4) + lrow) * DD + (hi << 3);
#pragma unroll
    for (int j = 0; j < 16; ++j) {
      kb[g][0][j] = kr[j < 8 ? j : j + 8];
      kb[g][1][j] = kr[32 + (j < 8 ? j : j + 8)];
    }
  }

  v8f of[2][4];
#pragma unroll
  for (int g = 0; g < 2; ++g)
#pragma unroll
    for (int i = 0; i < 4; ++i) of[g][i] = (v8f){};
  float m[2] = {-1e30f, -1e30f};
  float l[2] = {0.0f, 0.0f};
  const int tcol[2] = {qt0 + lrow, qt0 + 16 + lrow};

  const int nsteps = (qt0 >> 5) + 1;           // causal: s <= t, 32 keys/step
  for (int step = 0; step < nsteps; ++step) {
    const int s0 = step << 5;

    // Shared A operands: Q rows s0..s0+15 (tile0) and s0+16..s0+31 (tile1).
    v16h a0, a1, a2, a3;
    {
      const _Float16* q0r = Qb + (size_t)(s0 + lrow) * DD;
      const _Float16* q1r = Qb + (size_t)(s0 + 16 + lrow) * DD;
#pragma unroll
      for (int j = 0; j < 8; ++j) {
        a0[j] = q0r[k0a + j];      a0[8 + j] = q0r[16 + k0a + j];
        a1[j] = q0r[32 + k0a + j]; a1[8 + j] = q0r[48 + k0a + j];
        a2[j] = q1r[k0a + j];      a2[8 + j] = q1r[16 + k0a + j];
        a3[j] = q1r[32 + k0a + j]; a3[8 + j] = q1r[48 + k0a + j];
      }
    }

    v16h pa[2];
#pragma unroll
    for (int g = 0; g < 2; ++g) {
      v8f z0 = {}, z1 = {};
      z0 = wmma_f16(a0, kb[g][0], z0); z0 = wmma_f16(a1, kb[g][1], z0);
      z1 = wmma_f16(a2, kb[g][0], z1); z1 = wmma_f16(a3, kb[g][1], z1);

      // Causal mask: fires only on the last (diagonal) step, wave-uniform.
      if (s0 + 31 > qt0 + (g << 4)) {
        const int sb = s0 + rb;
#pragma unroll
        for (int v = 0; v < 8; ++v) {
          z0[v] = (sb + v      > tcol[g]) ? -1e30f : z0[v];
          z1[v] = (sb + 16 + v > tcol[g]) ? -1e30f : z1[v];
        }
      }

      // Online softmax over s for each query t (per-lane + xor16 combine).
      float mx = z0[0];
#pragma unroll
      for (int v = 0; v < 8; ++v) {
        mx = fmaxf(mx, z0[v]);
        mx = fmaxf(mx, z1[v]);
      }
      mx = fmaxf(mx, __shfl_xor(mx, 16, 32));
      const float newm  = fmaxf(m[g], mx);
      const float alpha = __expf(m[g] - newm);
      m[g] = newm;

      v8f p0, p1;
      float rs = 0.0f;
#pragma unroll
      for (int v = 0; v < 8; ++v) {
        p0[v] = __expf(z0[v] - newm);
        p1[v] = __expf(z1[v] - newm);
        rs += p0[v] + p1[v];
      }
      rs += __shfl_xor(rs, 16, 32);
      l[g] = l[g] * alpha + rs;

      // Rescale O accumulators: row r's alpha lives in lanes r and r+16.
#pragma unroll
      for (int v = 0; v < 8; ++v) {
        const float av = __shfl(alpha, rb + v, 32);
        of[g][0][v] *= av; of[g][1][v] *= av;
        of[g][2][v] *= av; of[g][3][v] *= av;
      }

      // Pack P^T (16 x 32 f16 A-matrix): identical lane layout to (z0,z1).
#pragma unroll
      for (int j = 0; j < 8; ++j) {
        pa[g][j]     = (_Float16)p0[j];
        pa[g][8 + j] = (_Float16)p1[j];
      }
    }

    // O += P^T x V : four 16-wide column tiles, V loads shared by both groups.
#pragma unroll
    for (int d0 = 0; d0 < 4; ++d0) {
      const _Float16* vr =
          Vb + (size_t)((d0 << 4) + lrow) * TT + s0 + (hi << 3);
      v16h bv;
#pragma unroll
      for (int j = 0; j < 16; ++j) bv[j] = vr[j < 8 ? j : j + 8];
      of[0][d0] = wmma_f16(pa[0], bv, of[0][d0]);
      of[1][d0] = wmma_f16(pa[1], bv, of[1][d0]);
    }
  }

  // Normalize by row sum l (broadcast like alpha) and store fp32 output.
  float* ob = out + (size_t)b * TT * DD;
#pragma unroll
  for (int g = 0; g < 2; ++g) {
#pragma unroll
    for (int v = 0; v < 8; ++v) {
      const float lv  = __shfl(l[g], rb + v, 32);
      const float inv = 1.0f / lv;
      const size_t roff = (size_t)(qt0 + (g << 4) + rb + v) * DD + lrow;
      ob[roff]      = of[g][0][v] * inv;
      ob[roff + 16] = of[g][1][v] * inv;
      ob[roff + 32] = of[g][2][v] * inv;
      ob[roff + 48] = of[g][3][v] * inv;
    }
  }
}

// ---------------------------------------------------------------------------
extern "C" void kernel_launch(void* const* d_in, const int* in_sizes, int n_in,
                              void* d_out, int out_size, void* d_ws,
                              size_t ws_size, hipStream_t stream) {
  const float* x  = (const float*)d_in[0];
  const float* Wk = (const float*)d_in[1];
  const float* Wq = (const float*)d_in[2];
  const float* Wv = (const float*)d_in[3];
  float* out = (float*)d_out;

  const size_t nelem = (size_t)BB * TT * DD;       // 4,194,304
  _Float16* kp = (_Float16*)d_ws;                  // 8 MB
  _Float16* qp = kp + nelem;                       // 8 MB
  _Float16* vt = qp + nelem;                       // 8 MB (transposed V)

  const dim3 block(256);                           // 8 waves / block

  const int ptiles = BB * TT / 16;                 // 4096 projection tiles
  proj_kernel<<<dim3(ptiles / 8), block, 0, stream>>>(x, Wk, Wq, Wv,
                                                      kp, qp, vt);

  const int atiles = BB * TT / 32;                 // 2048 attention tiles
  attn_kernel<<<dim3(atiles / 8), block, 0, stream>>>(kp, qp, vt, out);
}
